// SimpleVQAutoEncoder_7945689497873
// MI455X (gfx1250) — compile-verified, matrix operand order checked by reading, not run on hardware
//
#include <hip/hip_runtime.h>
#include <hip/hip_bf16.h>
#include <math.h>

typedef __attribute__((ext_vector_type(16))) _Float16 v16h;
typedef __attribute__((ext_vector_type(8)))  _Float16 v8h;
typedef __attribute__((ext_vector_type(8)))  float    v8f;

#define BATCH   64
#define HFULL   256
#define WFULL   256
#define HQ      64                      // quarter-res spatial
#define C2      32                      // latent channels
#define NROWS   (BATCH*HQ*HQ)           // 262144 VQ rows
#define IMG_ELEMS (BATCH*HFULL*WFULL)   // 4194304
#define IDX_OFF  IMG_ELEMS
#define LOSS_OFF (IMG_ELEMS + NROWS)

__device__ __forceinline__ float gelu_exact(float x) {
    return 0.5f * x * (1.0f + erff(x * 0.70710678118654752f));
}

__device__ __forceinline__ v16h pack16(v8h lo, v8h hi) {
    return __builtin_shufflevector(lo, hi, 0,1,2,3,4,5,6,7,8,9,10,11,12,13,14,15);
}

// ---------------------------------------------------------------------------
// Kernel 1: fused conv1(1->16) + maxpool2 + gelu + conv2(16->32, WMMA) + pool
// Block = one (image, 8x8 quarter-res tile). Output h in NHWC [B,64,64,32].
// conv2 as implicit GEMM: M=256 positions, N=32 oc, K=144 (padded 160).
// k-order = (tap, ci) so every 8-half A group is one contiguous LDS read.
// ---------------------------------------------------------------------------
__global__ __launch_bounds__(256)
void vqae_encoder(const float* __restrict__ x,
                  const float* __restrict__ w1, const float* __restrict__ b1,
                  const float* __restrict__ w2, const float* __restrict__ b2,
                  float* __restrict__ h_out)
{
    __shared__ float sX[38*38];                      // full-res input + halo
    __shared__ alignas(16) _Float16 sG[18*18*16];    // gelu(pool(conv1)) f16
    __shared__ float sW1[144], sB1[16], sB2s[32];
    __shared__ v16h  sBW2[5*2*32];                   // conv2 weights, B layout
    __shared__ float sConv[256*32];                  // conv2 raw outputs

    const int tid  = threadIdx.x;
    const int lane = tid & 31;
    const int wv   = tid >> 5;
    const int hi   = lane >> 4;
    const int khi  = hi << 3;
    const int b    = blockIdx.x >> 6;
    const int tile = blockIdx.x & 63;
    const int ty   = tile >> 3, tx = tile & 7;
    const int y0q  = ty * 8,  x0q = tx * 8;          // quarter-res origin

    for (int i = tid; i < 144; i += 256) sW1[i] = w1[i];
    if (tid < 16) sB1[tid] = b1[tid];
    if (tid < 32) sB2s[tid] = b2[tid];

    // conv2 weights -> LDS in B-matrix layout (K padded 144 -> 160).
    for (int i = tid; i < 160*32; i += 256) {
        int k  = i >> 5;                 // 0..159
        int oc = i & 31;
        float v = 0.f;
        if (k < 144) {
            int tap = k >> 4, ci = k & 15;
            v = w2[(oc*16 + ci)*9 + tap];
        }
        int cch  = k >> 5;               // k-chunk 0..4
        int krel = k & 31;
        int nt = oc >> 4, n = oc & 15;
        int ln = n + (((krel >> 3) & 1) << 4);
        int hh = (krel & 7) + ((krel >> 4) << 3);
        ((_Float16*)sBW2)[(((cch*2 + nt)*32 + ln) << 4) + hh] = (_Float16)v;
    }

    // Load x tile: full-res rows [32*ty-3, 32*ty+34], zero-padded.
    const int yx0 = 32*ty - 3, xx0 = 32*tx - 3;
    const float* xb = x + (size_t)b * HFULL * WFULL;
    for (int i = tid; i < 38*38; i += 256) {
        int r = i / 38, c = i % 38;
        int yy = yx0 + r, xx = xx0 + c;
        float v = 0.f;
        if (yy >= 0 && yy < HFULL && xx >= 0 && xx < WFULL) v = xb[yy*WFULL + xx];
        sX[i] = v;
    }
    __syncthreads();

    // Stage 1: conv1 + maxpool2 + gelu at 18x18 half-res positions (f16 out).
    for (int p = tid; p < 18*18; p += 256) {
        int py = p / 18, px = p % 18;
        int yh = 16*ty - 1 + py;
        int xh = 16*tx - 1 + px;
        bool inb = (yh >= 0 && yh < 128 && xh >= 0 && xh < 128);
        float m[16];
        #pragma unroll
        for (int ch = 0; ch < 16; ++ch) m[ch] = -3.0e38f;
        if (inb) {
            #pragma unroll
            for (int sub = 0; sub < 4; ++sub) {
                int r = sub >> 1, c = sub & 1;
                int ry = 2*py + r + 1;
                int rx = 2*px + c + 1;
                float v[9];
                #pragma unroll
                for (int dy = 0; dy < 3; ++dy)
                    #pragma unroll
                    for (int dx = 0; dx < 3; ++dx)
                        v[dy*3+dx] = sX[(ry-1+dy)*38 + (rx-1+dx)];
                #pragma unroll
                for (int ch = 0; ch < 16; ++ch) {
                    const float* w = &sW1[ch*9];
                    float s = 0.f;
                    #pragma unroll
                    for (int k = 0; k < 9; ++k) s += w[k] * v[k];
                    m[ch] = fmaxf(m[ch], s);
                }
            }
        }
        #pragma unroll
        for (int ch = 0; ch < 16; ++ch)
            sG[p*16 + ch] = (_Float16)(inb ? gelu_exact(m[ch] + sB1[ch]) : 0.f);
    }
    __syncthreads();

    // Stage 2: conv2 via WMMA. 16 M-tiles x 2 N-tiles x 5 k-chunks.
    for (int mt = wv; mt < 16; mt += 8) {
        const int p   = mt*16 + (lane & 15);   // this lane's A row (position)
        const int py2 = p >> 4, px2 = p & 15;  // conv2 position in 16x16 grid
        v16h A[5];
        #pragma unroll
        for (int c = 0; c < 5; ++c) {
            const int t0 = 2*c, t1 = 2*c + 1;
            v8h lo = *(const v8h*)&sG[((py2 + t0/3)*18 + (px2 + t0%3))*16 + khi];
            v8h hv = {};
            if (t1 < 9)
                hv = *(const v8h*)&sG[((py2 + t1/3)*18 + (px2 + t1%3))*16 + khi];
            A[c] = pack16(lo, hv);
        }
        #pragma unroll
        for (int nt = 0; nt < 2; ++nt) {
            v8f cacc = {};
            #pragma unroll
            for (int c = 0; c < 5; ++c)
                cacc = __builtin_amdgcn_wmma_f32_16x16x32_f16(
                           false, A[c], false, sBW2[(c*2 + nt)*32 + lane],
                           (short)0, cacc, false, false);
            const int oc = nt*16 + (lane & 15);
            #pragma unroll
            for (int j = 0; j < 8; ++j)
                sConv[(mt*16 + j + (hi << 3))*32 + oc] = cacc[j];
        }
    }
    __syncthreads();

    // Stage 3: maxpool2 + bias -> 8x8 quarter-res outputs (NHWC).
    for (int o = tid; o < 8*8*32; o += 256) {
        int oc  = o & 31;
        int pos = o >> 5;
        int qy = pos >> 3, qx = pos & 7;
        float a0 = sConv[((2*qy+0)*16 + 2*qx+0)*32 + oc];
        float a1 = sConv[((2*qy+0)*16 + 2*qx+1)*32 + oc];
        float a2 = sConv[((2*qy+1)*16 + 2*qx+0)*32 + oc];
        float a3 = sConv[((2*qy+1)*16 + 2*qx+1)*32 + oc];
        float best = fmaxf(fmaxf(a0, a1), fmaxf(a2, a3));
        h_out[(((size_t)b*HQ + (y0q+qy))*HQ + (x0q+qx))*C2 + oc] = best + sB2s[oc];
    }
}

// ---------------------------------------------------------------------------
// Kernel 2: vector quantize via WMMA.  d = |x|^2 - 2 x.c + |c|^2, argmin.
// ---------------------------------------------------------------------------
__global__ __launch_bounds__(256)
void vqae_vq(const float* __restrict__ h,
             const float* __restrict__ codebook,
             float* __restrict__ idx_out,
             float* __restrict__ loss_partial)
{
    __shared__ v16h  sB[32*32];        // 32 code tiles, B-matrix layout, 32KB
    __shared__ float sCN[512];
    __shared__ float sRN[8*16];
    __shared__ float sLoss[256];

    const int tid  = threadIdx.x;
    const int lane = tid & 31;
    const int wv   = tid >> 5;
    const int m    = lane & 15;
    const int hi   = lane >> 4;

    _Float16* sBh = (_Float16*)sB;
    for (int i = tid; i < 512*32; i += 256) {
        int code = i >> 5, ch = i & 31;
        int t  = code >> 4, n = code & 15;
        int bl = n + (((ch >> 3) & 1) << 4);
        int hh = (ch & 7) + ((ch >> 4) << 3);
        sBh[((t*32 + bl) << 4) + hh] = (_Float16)codebook[i];
    }
    for (int code = tid; code < 512; code += 256) {
        const float* cp = codebook + code*32;
        float s = 0.f;
        for (int c2 = 0; c2 < 32; ++c2) s += cp[c2]*cp[c2];
        sCN[code] = s;
    }

    const int rbase = blockIdx.x*128 + wv*16;
    const float* rp = h + (size_t)(rbase + m)*32;
    const int khi = hi ? 8 : 0;
    v16h a;
    float rnp = 0.f;
    #pragma unroll
    for (int i = 0; i < 8; ++i) { float v = rp[khi + i];      a[i]   = (_Float16)v; rnp += v*v; }
    #pragma unroll
    for (int i = 0; i < 8; ++i) { float v = rp[16 + khi + i]; a[8+i] = (_Float16)v; rnp += v*v; }
    float rn = rnp + __shfl_xor(rnp, 16, 32);
    if (hi == 0) sRN[wv*16 + m] = rn;
    __syncthreads();

    float bestd[8];
    int   besti[8];
    #pragma unroll
    for (int j = 0; j < 8; ++j) { bestd[j] = 3.0e38f; besti[j] = 0; }

    const int rowoff = hi << 3;
    #pragma unroll 2
    for (int t = 0; t < 32; ++t) {
        v16h bt = sB[t*32 + lane];
        v8f  c  = {};
        c = __builtin_amdgcn_wmma_f32_16x16x32_f16(false, a, false, bt,
                                                   (short)0, c, false, false);
        int code = (t << 4) + m;
        float cn = sCN[code];
        #pragma unroll
        for (int j = 0; j < 8; ++j) {
            float d = sRN[wv*16 + j + rowoff] - 2.0f*c[j] + cn;
            if (d < bestd[j]) { bestd[j] = d; besti[j] = code; }
        }
    }

    #pragma unroll
    for (int mk = 1; mk <= 8; mk <<= 1) {
        #pragma unroll
        for (int j = 0; j < 8; ++j) {
            float od = __shfl_xor(bestd[j], mk, 32);
            int   oi = __shfl_xor(besti[j], mk, 32);
            if (od < bestd[j] || (od == bestd[j] && oi < besti[j])) {
                bestd[j] = od; besti[j] = oi;
            }
        }
    }

    float lsum = 0.f;
    if (m == 0) {
        #pragma unroll 1
        for (int j = 0; j < 8; ++j) {
            int r  = rbase + j + rowoff;
            int ix = besti[j];
            idx_out[r] = (float)ix;
            const float* cp = codebook + (size_t)ix*32;
            const float* fp = h + (size_t)r*32;
            for (int c2 = 0; c2 < 32; ++c2) {
                float dd = cp[c2] - fp[c2];
                lsum += dd*dd;
            }
        }
    }
    sLoss[tid] = lsum;
    __syncthreads();
    for (int s = 128; s > 0; s >>= 1) {
        if (tid < s) sLoss[tid] += sLoss[tid + s];
        __syncthreads();
    }
    if (tid == 0) loss_partial[blockIdx.x] = sLoss[0];
}

// ---------------------------------------------------------------------------
// Kernel 3: deterministic commit-loss reduction.
// ---------------------------------------------------------------------------
__global__ __launch_bounds__(256)
void vqae_loss_reduce(const float* __restrict__ partial, int nblocks,
                      float* __restrict__ loss_out)
{
    __shared__ float s[256];
    float acc = 0.f;
    for (int i = threadIdx.x; i < nblocks; i += 256) acc += partial[i];
    s[threadIdx.x] = acc;
    __syncthreads();
    for (int st = 128; st > 0; st >>= 1) {
        if (threadIdx.x < st) s[threadIdx.x] += s[threadIdx.x + st];
        __syncthreads();
    }
    if (threadIdx.x == 0) loss_out[0] = s[0] / (float)((size_t)NROWS * 32);
}

// ---------------------------------------------------------------------------
// Kernel 4: fused up2 + conv3(32->16, WMMA) + gelu + up2 + conv4(16->1) + clip
// conv3 as implicit GEMM: M=324 positions (pad 336), N=16 oc, K=288 (9 chunks,
// one chunk == one tap's 32 channels -> A groups are contiguous LDS rows).
// ---------------------------------------------------------------------------
__global__ __launch_bounds__(256)
void vqae_decoder(const float* __restrict__ idx_f,
                  const float* __restrict__ codebook,
                  const float* __restrict__ w3, const float* __restrict__ b3,
                  const float* __restrict__ w4, const float* __restrict__ b4,
                  float* __restrict__ out)
{
    __shared__ alignas(16) _Float16 sU1[20*20*32];   // up2(q) half-res f16
    __shared__ v16h  sBW3[9*32];                     // conv3 weights, B layout
    __shared__ float sG3[324*16];                    // gelu(conv3) f32
    __shared__ float sB3s[16], sW4s[144];
    __shared__ float sB4s;

    const int tid  = threadIdx.x;
    const int lane = tid & 31;
    const int wv   = tid >> 5;
    const int hi   = lane >> 4;
    const int khi  = hi << 3;
    const int b    = blockIdx.x >> 6;
    const int tile = blockIdx.x & 63;
    const int ty   = tile >> 3, tx = tile & 7;
    const int y0   = ty * 32, x0 = tx * 32;

    for (int i = tid; i < 144; i += 256) sW4s[i] = w4[i];
    if (tid < 16) sB3s[tid] = b3[tid];
    if (tid == 0) sB4s = b4[0];

    // conv3 weights -> LDS in B-matrix layout (K = 9 chunks of 32).
    for (int i = tid; i < 288*16; i += 256) {
        int k  = i >> 4;                 // 0..287
        int oc = i & 15;
        int tap = k >> 5, ci = k & 31;
        float v = w3[(oc*32 + ci)*9 + tap];
        int ln = oc + (((ci >> 3) & 1) << 4);
        int hh = (ci & 7) + ((ci >> 4) << 3);
        ((_Float16*)sBW3)[((tap*32 + ln) << 4) + hh] = (_Float16)v;
    }

    // Build u1 = up2(q) half-res tile 20x20x32 (f16), gathered from codebook.
    const int u1y0 = 16*ty - 2, u1x0 = 16*tx - 2;    // half-res origin of sU1
    for (int i = tid; i < 20*20*32; i += 256) {
        int ch  = i & 31;
        int pos = i >> 5;
        int r = pos / 20, c = pos % 20;
        int yh = u1y0 + r, xh = u1x0 + c;
        int yq = yh >> 1, xq = xh >> 1;              // arithmetic shift = floor
        float v = 0.f;
        if (yq >= 0 && yq < HQ && xq >= 0 && xq < HQ) {
            int ix = (int)idx_f[((size_t)b*HQ + yq)*HQ + xq];
            v = codebook[ix*32 + ch];
        }
        sU1[pos*32 + ch] = (_Float16)v;
    }
    __syncthreads();

    // conv3 via WMMA + bias + gelu -> sG3. 21 M-tiles x 9 k-chunks.
    for (int mt = wv; mt < 21; mt += 8) {
        int p  = mt*16 + (lane & 15);
        int pc = p < 324 ? p : 323;                  // clamp pad rows
        int py = pc / 18, px = pc % 18;              // conv3 position
        v8f cacc = {};
        #pragma unroll
        for (int t = 0; t < 9; ++t) {
            const _Float16* up = &sU1[((py + t/3)*20 + (px + t%3))*32 + khi];
            v8h lo = *(const v8h*)up;
            v8h hv = *(const v8h*)(up + 16);
            cacc = __builtin_amdgcn_wmma_f32_16x16x32_f16(
                       false, pack16(lo, hv), false, sBW3[t*32 + lane],
                       (short)0, cacc, false, false);
        }
        const int oc = lane & 15;
        #pragma unroll
        for (int j = 0; j < 8; ++j) {
            int pos = mt*16 + j + (hi << 3);
            if (pos < 324)
                sG3[pos*16 + oc] = gelu_exact(cacc[j] + sB3s[oc]);
        }
    }
    __syncthreads();

    // conv4 + clip at 32x32 full-res pixels (u2 = nearest-up2 of g3).
    for (int p = tid; p < 32*32; p += 256) {
        int py = p >> 5, px = p & 31;
        float s = sB4s;
        #pragma unroll
        for (int dy = -1; dy <= 1; ++dy)
            #pragma unroll
            for (int dx = -1; dx <= 1; ++dx) {
                int rr = (py + dy + 2) >> 1;
                int cc = (px + dx + 2) >> 1;
                const float* gp = &sG3[(rr*18 + cc)*16];
                const int woff = (dy+1)*3 + (dx+1);
                #pragma unroll
                for (int ci = 0; ci < 16; ++ci)
                    s += sW4s[ci*9 + woff] * gp[ci];
            }
        s = fminf(1.0f, fmaxf(-1.0f, s));
        out[((size_t)b*HFULL + (y0+py))*WFULL + (x0+px)] = s;
    }
}

// ---------------------------------------------------------------------------
extern "C" void kernel_launch(void* const* d_in, const int* in_sizes, int n_in,
                              void* d_out, int out_size, void* d_ws, size_t ws_size,
                              hipStream_t stream)
{
    const float* x  = (const float*)d_in[0];
    const float* w1 = (const float*)d_in[1];
    const float* b1 = (const float*)d_in[2];
    const float* w2 = (const float*)d_in[3];
    const float* b2 = (const float*)d_in[4];
    const float* cb = (const float*)d_in[5];
    const float* w3 = (const float*)d_in[6];
    const float* b3 = (const float*)d_in[7];
    const float* w4 = (const float*)d_in[8];
    const float* b4 = (const float*)d_in[9];

    float* out  = (float*)d_out;
    float* h    = (float*)d_ws;                     // NROWS*32 floats = 32MB
    float* part = h + (size_t)NROWS * 32;           // 2048 floats

    vqae_encoder<<<BATCH*64, 256, 0, stream>>>(x, w1, b1, w2, b2, h);
    vqae_vq<<<NROWS/128, 256, 0, stream>>>(h, cb, out + IDX_OFF, part);
    vqae_loss_reduce<<<1, 256, 0, stream>>>(part, NROWS/128, out + LOSS_OFF);
    vqae_decoder<<<BATCH*64, 256, 0, stream>>>(out + IDX_OFF, cb, w3, b3, w4, b4, out);
}
